// TransformerLayer_68899865362783
// MI455X (gfx1250) — matrix operand order repaired and checked by reading, not compiled, not run on hardware
//
#include <hip/hip_runtime.h>
#include <hip/hip_bf16.h>
#include <stdint.h>

// ---- problem constants (match reference) ----
#define B_   2
#define T_   2048
#define E_   2048
#define H_   16
#define DH_  128
#define FFN_ 8192

typedef unsigned short u16;
typedef __attribute__((ext_vector_type(16))) __bf16 v16bf;
typedef __attribute__((ext_vector_type(8)))  float  v8f;
typedef __attribute__((ext_vector_type(4)))  unsigned v4u;
typedef __attribute__((ext_vector_type(8)))  int      v8i;
typedef __attribute__((ext_vector_type(4)))  int      v4i;

union Frag { uint4 q[2]; v16bf v; };

__device__ __forceinline__ float bf2f(u16 u) {
    union { unsigned i; float f; } c; c.i = ((unsigned)u) << 16; return c.f;
}
__device__ __forceinline__ u16 f2bf(float f) {
    union { float f; unsigned i; } c; c.f = f;
    unsigned r = c.i + 0x7fffu + ((c.i >> 16) & 1u);   // RNE
    return (u16)(r >> 16);
}

// ---------------- Tensor Data Mover (TDM) 2D tile load ----------------
#if defined(__has_builtin)
#if __has_builtin(__builtin_amdgcn_tensor_load_to_lds)
#define HAVE_TDM 1
#endif
#endif
#ifndef HAVE_TDM
#define HAVE_TDM 0
#endif

#if HAVE_TDM
// Load a tile_h x tile_w (u16 elements) tile from global (row stride row_stride
// elements) into LDS at lds_off, inserting pad bytes per D# pad fields.
// pad_interval_code: pad after 2^(c+1) dwords; pad_amount_code: (c+1) dwords pad.
__device__ __forceinline__ void tdm_load_2d_u16(unsigned lds_off, const void* gptr,
                                                unsigned tile_w, unsigned tile_h,
                                                unsigned long long row_stride,
                                                unsigned pad_interval_code,
                                                unsigned pad_amount_code,
                                                unsigned pad_en) {
    unsigned long long ga = (unsigned long long)(uintptr_t)gptr;
    v4u g0;
    g0.x = 1u;                                            // count=1, user mode
    g0.y = lds_off;                                       // lds_addr
    g0.z = (unsigned)(ga & 0xffffffffu);                  // global_addr[31:0]
    g0.w = (unsigned)((ga >> 32) & 0x1ffffffu) | (2u << 30);  // ga[56:32], type=2
    const unsigned long long td0 = row_stride;            // tensor_dim0 (elems)
    const unsigned long long td1 = 0x7fffffffull;         // tensor_dim1 (rows, generous)
    const unsigned long long s0  = row_stride;            // tensor_dim0_stride (elems)
    v8i g1;
    g1[0] = (int)((1u << 16) |                            // data_size = 1 (2 bytes)
                  (pad_en << 20) | (pad_interval_code << 22) | (pad_amount_code << 25));
    g1[1] = (int)((unsigned)(td0 & 0xffffu) << 16);       // ab_addr=0 | td0[15:0]
    g1[2] = (int)(((td0 >> 16) & 0xffffu) | ((td1 & 0xffffu) << 16));
    g1[3] = (int)(((td1 >> 16) & 0xffffu) | (tile_w << 16));   // tile_dim0
    g1[4] = (int)(tile_h & 0xffffu);                      // tile_dim1 | tile_dim2=0
    g1[5] = (int)(s0 & 0xffffffffu);
    g1[6] = (int)((s0 >> 32) & 0xffffu);                  // | dim1_stride[15:0]=0
    g1[7] = 0;
    v4i gz4 = {0, 0, 0, 0};
#if __clang_major__ >= 23
    v8i gz8 = {0, 0, 0, 0, 0, 0, 0, 0};
    __builtin_amdgcn_tensor_load_to_lds(g0, g1, gz4, gz4, gz8, 0);
#else
    __builtin_amdgcn_tensor_load_to_lds(g0, g1, gz4, gz4, 0);
#endif
}
#endif

// ---------------- fp32 -> bf16 cast ----------------
__global__ __launch_bounds__(256) void k_cast_bf16(const float* __restrict__ src,
                                                   u16* __restrict__ dst, int n) {
    int i = blockIdx.x * 256 + threadIdx.x;
    if (i < n) dst[i] = f2bf(src[i]);
}

// ---------------- LayerNorm (one row of E=2048 per block) ----------------
__global__ __launch_bounds__(256) void k_layernorm(const float* __restrict__ x,
                                                   const float* __restrict__ g,
                                                   const float* __restrict__ b,
                                                   u16* __restrict__ out) {
    __shared__ float s1[256], s2[256];
    const int row = blockIdx.x, tid = threadIdx.x;
    const float* xr = x + (size_t)row * E_;
    float v[8]; float sum = 0.f, sq = 0.f;
#pragma unroll
    for (int i = 0; i < 8; ++i) { float t = xr[tid + i * 256]; v[i] = t; sum += t; sq += t * t; }
    s1[tid] = sum; s2[tid] = sq; __syncthreads();
    for (int off = 128; off > 0; off >>= 1) {
        if (tid < off) { s1[tid] += s1[tid + off]; s2[tid] += s2[tid + off]; }
        __syncthreads();
    }
    const float mu = s1[0] * (1.0f / E_);
    const float var = s2[0] * (1.0f / E_) - mu * mu;
    const float rstd = rsqrtf(var + 1e-5f);
    u16* orow = out + (size_t)row * E_;
#pragma unroll
    for (int i = 0; i < 8; ++i) {
        int c = tid + i * 256;
        orow[c] = f2bf((v[i] - mu) * rstd * g[c] + b[c]);
    }
}

// ---------------- tiled WMMA GEMM: C[MxN] = A[MxK] * W[KxN] ----------------
// MODE 0: bf16 output.  MODE 1: f32 output with residual add.
// Block tile 128x128, K-step 32, 8 waves (4x2), wave tile 32x64 (2x4 WMMA tiles).
// A-tile staged by TDM (wave 0, TENSORcnt); B-tile staged transposed by all threads.
template <int MODE>
__global__ __launch_bounds__(256) void k_gemm(const u16* __restrict__ A,
                                              const u16* __restrict__ W,
                                              const float* __restrict__ resid,
                                              void* __restrict__ Cout,
                                              int M, int N, int K) {
    __shared__ __align__(16) u16 As[128][40];   // [m][k], 16B pad per 64B row
    __shared__ __align__(16) u16 Bs[128][40];   // [n][k]  (transposed on stage)
    const int tid = threadIdx.x;
    const int lane = tid & 31, wave = tid >> 5;
    const int wm = (wave & 3) * 32;
    const int wn = (wave >> 2) * 64;
    const int bm = blockIdx.y * 128;
    const int bn = blockIdx.x * 128;

    v8f acc[2][4] = {};
    const int nk = K >> 5;

    for (int kb = 0; kb < nk; ++kb) {
#if HAVE_TDM
        // A tile 128x32 via Tensor Data Mover: row = 64B -> pad interval code 3,
        // LDS row pad 16B = 4 dwords -> amount code 3.
        if (wave == 0) {
            tdm_load_2d_u16((unsigned)(uintptr_t)&As[0][0],
                            A + (size_t)bm * K + kb * 32,
                            32u, 128u, (unsigned long long)K, 3u, 3u, 1u);
        }
#else
#pragma unroll
        for (int i = 0; i < 2; ++i) {
            int idx = tid * 2 + i;
            int r = idx >> 2, c = (idx & 3) * 8;
            *(uint4*)&As[r][c] =
                *(const uint4*)(A + (size_t)(bm + r) * K + kb * 32 + c);
        }
#endif
        // stage B tile 32x128, transposed into Bs[n][k]
#pragma unroll
        for (int i = 0; i < 8; ++i) {
            int idx = i * 256 + tid;
            int kk = idx >> 6, np = (idx & 63) * 2;
            unsigned u = *(const unsigned*)(W + (size_t)(kb * 32 + kk) * N + bn + np);
            Bs[np][kk]     = (u16)(u & 0xffffu);
            Bs[np + 1][kk] = (u16)(u >> 16);
        }
        if (kb + 1 < nk) {
            __builtin_prefetch(W + (size_t)((kb + 1) * 32 + (tid >> 3)) * N + bn + (tid & 7) * 16, 0, 0);
        }
#if HAVE_TDM
        if (wave == 0) __builtin_amdgcn_s_wait_tensorcnt(0);
#endif
        __syncthreads();

        // fragments: ISA 16-bit A layout (K runs 0-7/16-23 | 8-15/24-31 per half-wave)
        Frag a[2], b[4];
        const int ar = lane & 15;
        const int ak = (lane < 16) ? 0 : 8;
#pragma unroll
        for (int mt = 0; mt < 2; ++mt) {
            int m = wm + mt * 16 + ar;
            a[mt].q[0] = *(const uint4*)&As[m][ak];
            a[mt].q[1] = *(const uint4*)&As[m][ak + 16];
        }
        const int bk = (lane < 16) ? 0 : 8;
#pragma unroll
        for (int nt = 0; nt < 4; ++nt) {
            int n = wn + nt * 16 + ar;
            b[nt].q[0] = *(const uint4*)&Bs[n][bk];
            b[nt].q[1] = *(const uint4*)&Bs[n][bk + 16];
        }
#pragma unroll
        for (int mt = 0; mt < 2; ++mt)
#pragma unroll
            for (int nt = 0; nt < 4; ++nt)
                acc[mt][nt] = __builtin_amdgcn_wmma_f32_16x16x32_bf16(
                    false, a[mt].v, false, b[nt].v, (short)0, acc[mt][nt], false, false);
        __syncthreads();
    }

    // epilogue: C layout — lane%16 = n, VGPR r -> m = r (+8 for upper half-wave)
    const int cn = lane & 15, half8 = ((lane >> 4) & 1) * 8;
#pragma unroll
    for (int mt = 0; mt < 2; ++mt)
#pragma unroll
        for (int nt = 0; nt < 4; ++nt) {
            int gn = bn + wn + nt * 16 + cn;
#pragma unroll
            for (int r = 0; r < 8; ++r) {
                int gm = bm + wm + mt * 16 + r + half8;
                float vv = acc[mt][nt][r];
                if (MODE == 0) {
                    ((u16*)Cout)[(size_t)gm * N + gn] = f2bf(vv);
                } else {
                    size_t o = (size_t)gm * N + gn;
                    ((float*)Cout)[o] = vv + resid[o];
                }
            }
        }
}

// ---------------- causal flash attention (bf16 WMMA, online softmax) ----------------
// grid: (T/128, H, B); 8 waves/block; each wave owns a 16-row Q tile.
__global__ __launch_bounds__(256) void k_attn(const u16* __restrict__ q,
                                              const u16* __restrict__ k,
                                              const u16* __restrict__ v,
                                              u16* __restrict__ ctx) {
    __shared__ __align__(16) u16 Ks[32][136];     // [key][d], 16B pad per 256B row
    __shared__ __align__(16) u16 Vt[128][40];     // [d][key] (transposed)
    __shared__ __align__(16) u16 Ps[8][16][40];   // per-wave P scratch [m][k]
    const int tid = threadIdx.x, lane = tid & 31, wave = tid >> 5;
    const int qt = blockIdx.x, h = blockIdx.y, bb = blockIdx.z;
    const int qbase = qt * 128 + wave * 16;
    const size_t headoff = (size_t)h * DH_;
    const int ar = lane & 15;
    const int half8 = ((lane >> 4) & 1) * 8;

    // preload Q fragments for this wave's 16 rows (K = 128 in 4 chunks of 32)
    Frag qa[4];
    {
        const u16* qrow = q + ((size_t)(bb * T_ + qbase + ar)) * E_ + headoff;
#pragma unroll
        for (int c = 0; c < 4; ++c) {
            int kb0 = c * 32 + ((lane < 16) ? 0 : 8);
            qa[c].q[0] = *(const uint4*)(qrow + kb0);
            qa[c].q[1] = *(const uint4*)(qrow + kb0 + 16);
        }
    }

    v8f acc[8] = {};
    float mx[8], li[8];
#pragma unroll
    for (int r = 0; r < 8; ++r) { mx[r] = -3e30f; li[r] = 0.f; }
    const float scale = 0.08838834764831845f;  // 1/sqrt(128)

    const int nkb = qt * 4 + 4;  // causal: keys [0, qt*128+128) in 32-key blocks
    for (int kb = 0; kb < nkb; ++kb) {
#if HAVE_TDM
        // K tile 32x128 via TDM: row = 256B -> interval code 5; pad 16B -> code 3.
        if (wave == 0) {
            tdm_load_2d_u16((unsigned)(uintptr_t)&Ks[0][0],
                            k + ((size_t)(bb * T_ + kb * 32)) * E_ + headoff,
                            128u, 32u, (unsigned long long)E_, 5u, 3u, 1u);
        }
#else
#pragma unroll
        for (int i = 0; i < 2; ++i) {
            int idx = tid * 2 + i;
            int key = idx >> 4, d = (idx & 15) * 8;
            *(uint4*)&Ks[key][d] =
                *(const uint4*)(k + ((size_t)(bb * T_ + kb * 32 + key)) * E_ + headoff + d);
        }
#endif
        // stage V tile transposed into Vt[d][key]
#pragma unroll
        for (int i = 0; i < 8; ++i) {
            int idx = i * 256 + tid;
            int key = idx >> 6, dp = (idx & 63) * 2;
            unsigned u = *(const unsigned*)(v + ((size_t)(bb * T_ + kb * 32 + key)) * E_ + headoff + dp);
            Vt[dp][key]     = (u16)(u & 0xffffu);
            Vt[dp + 1][key] = (u16)(u >> 16);
        }
#if HAVE_TDM
        if (wave == 0) __builtin_amdgcn_s_wait_tensorcnt(0);
#endif
        __syncthreads();

        // S = Q·K^T : two 16-key sub-tiles, 4 K-chunks each
        v8f s[2];
#pragma unroll
        for (int kt = 0; kt < 2; ++kt) {
            v8f sv = {};
            int key_l = kt * 16 + ar;
            int db = (lane < 16) ? 0 : 8;
#pragma unroll
            for (int c = 0; c < 4; ++c) {
                Frag bf;
                bf.q[0] = *(const uint4*)&Ks[key_l][c * 32 + db];
                bf.q[1] = *(const uint4*)&Ks[key_l][c * 32 + db + 16];
                sv = __builtin_amdgcn_wmma_f32_16x16x32_bf16(
                    false, qa[c].v, false, bf.v, (short)0, sv, false, false);
            }
            s[kt] = sv;
        }
        // scale + causal mask
#pragma unroll
        for (int kt = 0; kt < 2; ++kt) {
            int gk = kb * 32 + kt * 16 + ar;
#pragma unroll
            for (int r = 0; r < 8; ++r) {
                int gq = qbase + r + half8;
                float sv = s[kt][r] * scale;
                s[kt][r] = (gk <= gq) ? sv : -3e30f;
            }
        }
        // online softmax per row (butterfly over the 16-lane half)
#pragma unroll
        for (int r = 0; r < 8; ++r) {
            float m0 = fmaxf(s[0][r], s[1][r]);
#pragma unroll
            for (int off = 1; off < 16; off <<= 1) m0 = fmaxf(m0, __shfl_xor(m0, off, 32));
            float mn = fmaxf(mx[r], m0);
            float c0 = __expf(mx[r] - mn);
            float p0 = __expf(s[0][r] - mn);
            float p1 = __expf(s[1][r] - mn);
            s[0][r] = p0; s[1][r] = p1;
            float rs = p0 + p1;
#pragma unroll
            for (int off = 1; off < 16; off <<= 1) rs += __shfl_xor(rs, off, 32);
            li[r] = li[r] * c0 + rs;
            mx[r] = mn;
#pragma unroll
            for (int dt = 0; dt < 8; ++dt) acc[dt][r] *= c0;
        }
        // P (C-layout) -> LDS -> A-layout fragment
#pragma unroll
        for (int kt = 0; kt < 2; ++kt)
#pragma unroll
            for (int r = 0; r < 8; ++r)
                Ps[wave][r + half8][kt * 16 + ar] = f2bf(s[kt][r]);

        Frag pa;
        {
            int pk = (lane < 16) ? 0 : 8;
            pa.q[0] = *(const uint4*)&Ps[wave][ar][pk];
            pa.q[1] = *(const uint4*)&Ps[wave][ar][pk + 16];
        }
        // acc += P·V over 8 output d-tiles
        const int vk = (lane < 16) ? 0 : 8;
#pragma unroll
        for (int dt = 0; dt < 8; ++dt) {
            Frag bf;
            bf.q[0] = *(const uint4*)&Vt[dt * 16 + ar][vk];
            bf.q[1] = *(const uint4*)&Vt[dt * 16 + ar][vk + 16];
            acc[dt] = __builtin_amdgcn_wmma_f32_16x16x32_bf16(
                false, pa.v, false, bf.v, (short)0, acc[dt], false, false);
        }
        __syncthreads();
    }

    // normalize and store ctx (bf16, (B,T,H*DH) layout)
#pragma unroll
    for (int r = 0; r < 8; ++r) {
        float inv = 1.0f / li[r];
        int gq = qbase + r + half8;
        u16* crow = ctx + ((size_t)(bb * T_ + gq)) * E_ + headoff;
#pragma unroll
        for (int dt = 0; dt < 8; ++dt)
            crow[dt * 16 + ar] = f2bf(acc[dt][r] * inv);
    }
}

// ---------------- SwiGLU gate: u = silu(u) * g (in place on u) ----------------
__global__ __launch_bounds__(256) void k_silu_gate(u16* __restrict__ u,
                                                   const u16* __restrict__ g, int n) {
    int i = blockIdx.x * 256 + threadIdx.x;
    if (i < n) {
        float a = bf2f(u[i]);
        float b = bf2f(g[i]);
        float s = a / (1.f + __expf(-a));
        u[i] = f2bf(s * b);
    }
}

// ---------------- host-side orchestration ----------------
extern "C" void kernel_launch(void* const* d_in, const int* in_sizes, int n_in,
                              void* d_out, int out_size, void* d_ws, size_t ws_size,
                              hipStream_t stream) {
    (void)in_sizes; (void)n_in; (void)out_size; (void)ws_size;
    const float* x  = (const float*)d_in[0];
    const float* g1 = (const float*)d_in[1];
    const float* b1 = (const float*)d_in[2];
    const float* wq = (const float*)d_in[3];
    const float* wk = (const float*)d_in[4];
    const float* wv = (const float*)d_in[5];
    const float* wo = (const float*)d_in[6];
    const float* g2 = (const float*)d_in[7];
    const float* b2 = (const float*)d_in[8];
    const float* w1 = (const float*)d_in[9];
    const float* w2 = (const float*)d_in[10];
    const float* w3 = (const float*)d_in[11];

    const size_t EE  = (size_t)E_ * E_;
    const size_t EF  = (size_t)E_ * FFN_;
    const size_t BT  = (size_t)B_ * T_;
    const size_t BTE = BT * E_;
    const size_t BTF = BT * FFN_;

    char* base = (char*)d_ws; size_t off = 0;
    auto alloc = [&](size_t bytes) -> void* {
        void* r = base + off; off += (bytes + 255) & ~(size_t)255; return r;
    };
    u16* wqb  = (u16*)alloc(EE * 2);
    u16* wkb  = (u16*)alloc(EE * 2);
    u16* wvb  = (u16*)alloc(EE * 2);
    u16* wob  = (u16*)alloc(EE * 2);
    u16* w1b  = (u16*)alloc(EF * 2);
    u16* w2b  = (u16*)alloc(EF * 2);
    u16* w3b  = (u16*)alloc(EF * 2);
    u16* h_bf = (u16*)alloc(BTE * 2);   // LN1 output; reused for ctx
    u16* q_bf = (u16*)alloc(BTE * 2);   // Q; reused for LN2 output
    u16* k_bf = (u16*)alloc(BTE * 2);
    u16* v_bf = (u16*)alloc(BTE * 2);
    float* xmid = (float*)alloc(BTE * 4);
    u16* u_bf = (u16*)alloc(BTF * 2);   // w1 branch; gated written in place
    u16* g_bf = (u16*)alloc(BTF * 2);   // w2 branch
    u16* ctx_bf = h_bf;
    u16* h2_bf  = q_bf;

    auto cast = [&](const float* s, u16* d, size_t n) {
        k_cast_bf16<<<dim3((unsigned)((n + 255) / 256)), dim3(256), 0, stream>>>(s, d, (int)n);
    };
    cast(wq, wqb, EE); cast(wk, wkb, EE); cast(wv, wvb, EE); cast(wo, wob, EE);
    cast(w1, w1b, EF); cast(w2, w2b, EF); cast(w3, w3b, EF);

    // --- attention block ---
    k_layernorm<<<dim3((unsigned)BT), dim3(256), 0, stream>>>(x, g1, b1, h_bf);

    dim3 gEE(E_ / 128, (unsigned)(BT / 128));
    k_gemm<0><<<gEE, dim3(256), 0, stream>>>(h_bf, wqb, nullptr, q_bf, (int)BT, E_, E_);
    k_gemm<0><<<gEE, dim3(256), 0, stream>>>(h_bf, wkb, nullptr, k_bf, (int)BT, E_, E_);
    k_gemm<0><<<gEE, dim3(256), 0, stream>>>(h_bf, wvb, nullptr, v_bf, (int)BT, E_, E_);

    k_attn<<<dim3(T_ / 128, H_, B_), dim3(256), 0, stream>>>(q_bf, k_bf, v_bf, ctx_bf);

    k_gemm<1><<<gEE, dim3(256), 0, stream>>>(ctx_bf, wob, x, xmid, (int)BT, E_, E_);

    // --- SwiGLU MLP block ---
    k_layernorm<<<dim3((unsigned)BT), dim3(256), 0, stream>>>(xmid, g2, b2, h2_bf);

    dim3 gEF(FFN_ / 128, (unsigned)(BT / 128));
    k_gemm<0><<<gEF, dim3(256), 0, stream>>>(h2_bf, w1b, nullptr, u_bf, (int)BT, FFN_, E_);
    k_gemm<0><<<gEF, dim3(256), 0, stream>>>(h2_bf, w2b, nullptr, g_bf, (int)BT, FFN_, E_);

    k_silu_gate<<<dim3((unsigned)(BTF / 256)), dim3(256), 0, stream>>>(u_bf, g_bf, (int)BTF);

    dim3 gFE(E_ / 128, (unsigned)(BT / 128));
    k_gemm<1><<<gFE, dim3(256), 0, stream>>>(u_bf, w3b, xmid, d_out, (int)BT, E_, FFN_);
}
